// MeanAggregator_53145925321283
// MI455X (gfx1250) — matrix operand order repaired and checked by reading, not compiled, not run on hardware
//
#include <hip/hip_runtime.h>

typedef __attribute__((ext_vector_type(16))) _Float16 v16h;
typedef __attribute__((ext_vector_type(8)))  float    v8f;

#define D_IN   128
#define D_OUT  128
#define K_NEI  16
#define WAVES_PER_BLOCK 16   // 512 threads
#define THREADS (WAVES_PER_BLOCK * 32)

// ---------------------------------------------------------------------------
// Kernel 1: one-time convert W (f32, [128][128]) -> f16 into workspace.
// ---------------------------------------------------------------------------
__global__ void MeanAgg_w2h_kernel(const float* __restrict__ w,
                                   _Float16* __restrict__ w16) {
    int i = blockIdx.x * blockDim.x + threadIdx.x;   // 16384 elements total
    w16[i] = (_Float16)w[i];
}

// ---------------------------------------------------------------------------
// Kernel 2: one wave32 per batch element.
//   A tile  : gathered neighbors, 16(K_NEI) x 128(d_in), f16, built from
//             global loads directly in contiguous-K A-layout (each embed row read
//             exactly once, 2 lanes x 4x64B contiguous chunks per row).
//   B tiles : W^T slices read from LDS-staged f16 W (B[k][n] = W[n][k]);
//             all 4 K-step fragments preloaded per N-tile (1 dscnt wait).
//   C       : 8 f32 accumulators per N-tile; relu + mean in epilogue,
//             half-wave fold via ds_swizzle SWAPX16.
// ---------------------------------------------------------------------------
__launch_bounds__(THREADS, 1)
__global__ void MeanAgg_main_kernel(const float* __restrict__ embed,
                                    const _Float16* __restrict__ w16,
                                    const int* __restrict__ neigh_idx,
                                    float* __restrict__ out,
                                    int batch) {
    __shared__ _Float16 lw[D_OUT * D_IN];   // 32 KB: W in f16, row-major [o][d]

    // Stage W into LDS (uint4 = 8 halves per load).
    {
        const uint4* src = (const uint4*)w16;
        uint4* dst = (uint4*)lw;
        #pragma unroll
        for (int i = threadIdx.x; i < (D_OUT * D_IN) / 8; i += THREADS)
            dst[i] = src[i];
    }
    __syncthreads();

    const int lane    = threadIdx.x & 31;
    const int wave    = threadIdx.x >> 5;
    const int halfSel = lane >> 4;      // 0: lanes 0-15, 1: lanes 16-31
    const int m       = lane & 15;      // neighbor (M) index / N index

    const int b = blockIdx.x * WAVES_PER_BLOCK + wave;   // wave-uniform
    if (b >= batch) return;   // exact division in practice; wave-uniform anyway

    // Row index for this lane's neighbor (lanes 16-31 mirror 0-15).
    const int row = neigh_idx[b * K_NEI + m];
    const float* abase = embed + (long)row * D_IN;

    // -------- Load A fragments (4 K-steps of 32) in contiguous-K order -----------
    // lane<16 : K = kt*32+0..15
    // lane>=16: K = kt*32+16..31
    v16h afrag[4];
    #pragma unroll
    for (int kt = 0; kt < 4; ++kt) {
        const float* p = abase + kt * 32 + halfSel * 16;
        v16h a;
        #pragma unroll
        for (int t = 0; t < 8; ++t) a[t]     = (_Float16)p[t];       // chunk 0
        #pragma unroll
        for (int t = 0; t < 8; ++t) a[8 + t] = (_Float16)p[8 + t];   // chunk 1
        afrag[kt] = a;
    }

    // -------- 8 N-tiles x 4 K-steps of v_wmma_f32_16x16x32_f16 -------------
    const float inv_k = 1.0f / (float)K_NEI;
    #pragma unroll
    for (int nt = 0; nt < 8; ++nt) {
        // B fragment source: lane holds column n = nt*16+m,
        // K = kt*32 + halfSel*16 .. +15 (16 contiguous halves, 32B aligned).
        const _Float16* bp = lw + (nt * 16 + m) * D_IN + halfSel * 16;

        // Preload all 4 K-step fragments -> single dscnt wait per N-tile.
        v16h bfrag[4];
        #pragma unroll
        for (int kt = 0; kt < 4; ++kt)
            bfrag[kt] = *(const v16h*)(bp + kt * 32);

        v8f c = {};
        #pragma unroll
        for (int kt = 0; kt < 4; ++kt) {
            c = __builtin_amdgcn_wmma_f32_16x16x32_f16(
                    /*neg_a=*/false, afrag[kt],
                    /*neg_b=*/false, bfrag[kt],
                    /*c_mod=*/(short)0, c,
                    /*reuse_a=*/false, /*reuse_b=*/false);
        }

        // relu + partial mean over the 8 rows this lane holds.
        float s = 0.0f;
        #pragma unroll
        for (int j = 0; j < 8; ++j)
            s += fmaxf(c[j], 0.0f);

        // Fold the other half-wave (rows M vs M+8 live in lane ^ 16):
        // ds_swizzle SWAPX16: offset[14:10]=xor=0x10, offset[4:0]=and=0x1f.
        int sw = __builtin_amdgcn_ds_swizzle(__float_as_int(s), 0x401F);
        s += __int_as_float(sw);

        if (halfSel == 0)
            __builtin_nontemporal_store(s * inv_k,
                                        &out[(long)b * D_OUT + nt * 16 + m]);
    }
}

// ---------------------------------------------------------------------------
extern "C" void kernel_launch(void* const* d_in, const int* in_sizes, int n_in,
                              void* d_out, int out_size, void* d_ws, size_t ws_size,
                              hipStream_t stream) {
    const float* embed     = (const float*)d_in[0];   // [1e6, 128] f32
    const float* weight    = (const float*)d_in[1];   // [128, 128] f32
    const int*   neigh_idx = (const int*)d_in[2];     // [B, 16] i32
    float*       out       = (float*)d_out;           // [B, 128] f32

    const int batch = in_sizes[2] / K_NEI;
    _Float16* w16 = (_Float16*)d_ws;                  // 32 KB scratch

    // Convert W to f16 once per launch (deterministic, graph-safe).
    MeanAgg_w2h_kernel<<<(D_OUT * D_IN) / 256, 256, 0, stream>>>(weight, w16);

    const int blocks = (batch + WAVES_PER_BLOCK - 1) / WAVES_PER_BLOCK;
    MeanAgg_main_kernel<<<blocks, THREADS, 0, stream>>>(embed, w16, neigh_idx,
                                                        out, batch);
}
